// ANFISFirstLayer_78829829751006
// MI455X (gfx1250) — matrix operand order repaired and checked by reading, not compiled, not run on hardware
//
#include <hip/hip_runtime.h>
#include <hip/hip_bf16.h>
#include <math.h>

// Problem constants (from reference): B=32, T=512, F=128, R=8, O=128
#define N_TOT   (32 * 512)     // 16384 samples
#define FDIM    128
#define RDIM    8
#define ODIM    128
#define MTILE   64             // rows per workgroup in the GEMM kernel

typedef __attribute__((ext_vector_type(16))) _Float16 v16h;
typedef __attribute__((ext_vector_type(8)))  _Float16 v8h;
typedef __attribute__((ext_vector_type(8)))  float    v8f;
typedef __attribute__((ext_vector_type(4)))  int      v4i;

// Address-space cast helpers for the async-to-LDS builtin, whose parameters
// are (v4i addrspace(1)*, v4i addrspace(3)*, imm, imm). Generic LDS pointers
// carry the shared aperture in the high 32 bits, so the low 32 bits are the
// as3 offset; global generic pointers are numerically identical to as1.
#define AS1P(p) ((__attribute__((address_space(1))) v4i*)(unsigned long long)(uintptr_t)(p))
#define AS3P(p) ((__attribute__((address_space(3))) v4i*)(unsigned int)(uintptr_t)(p))

#if defined(__AMDGCN__) && __has_builtin(__builtin_amdgcn_global_load_async_to_lds_b128)
#define HAVE_ASYNC_LDS 1
#else
#define HAVE_ASYNC_LDS 0
#endif

__device__ __forceinline__ void wait_asynccnt0() {
#if defined(__AMDGCN__)
#if __has_builtin(__builtin_amdgcn_s_wait_asynccnt)
    __builtin_amdgcn_s_wait_asynccnt(0);
#else
    asm volatile("s_wait_asynccnt 0x0" ::: "memory");
#endif
#endif
}

// ---------------------------------------------------------------------------
// Kernel 0: W (R,F,O) f32  ->  WT (R,O,F) f16   (transpose + downconvert)
// ---------------------------------------------------------------------------
__global__ __launch_bounds__(256) void anfis_wt_kernel(
    const float* __restrict__ W, _Float16* __restrict__ WT)
{
    int idx = blockIdx.x * 256 + threadIdx.x;      // 0 .. R*O*F-1
    int r   = idx >> 14;
    int rem = idx & 16383;
    int o   = rem >> 7;
    int f   = rem & 127;
    WT[idx] = (_Float16)W[(r * FDIM + f) * ODIM + o];
}

// ---------------------------------------------------------------------------
// Kernel 1: normalized firing strengths fw (N,R) f32, plus Xh = f16(X).
// One thread per (n,r); this kernel already streams all of X, so the f16
// downconvert for the GEMM's A operand is fused here (each thread emits its
// 16-feature slice -> contiguous 32B stores per lane).
// ---------------------------------------------------------------------------
__global__ __launch_bounds__(256) void anfis_firing_kernel(
    const float* __restrict__ X, const float* __restrict__ centers,
    const float* __restrict__ widths, float* __restrict__ fw,
    _Float16* __restrict__ Xh)
{
    int idx = blockIdx.x * 256 + threadIdx.x;      // 0 .. N*R-1
    int n = idx >> 3;
    int r = idx & 7;

    const float* __restrict__ xr = X + n * FDIM;
    const float* __restrict__ cr = centers + r * FDIM;
    const float* __restrict__ wr = widths + r * FDIM;

    float sum = 0.0f;
#pragma unroll 8
    for (int f = 0; f < FDIM; ++f) {
        float s = fabsf(wr[f]) + 0.1f;
        float d = xr[f] - cr[f];
        float z = -(d * d) / (2.0f * s * s);
        float m = __expf(z);
        sum += __logf(m + 1e-10f);
    }
    float firing = __expf(sum);

    float tot = firing;
    tot += __shfl_xor(tot, 1, 8);
    tot += __shfl_xor(tot, 2, 8);
    tot += __shfl_xor(tot, 4, 8);
    fw[idx] = firing / (tot + 1e-8f);

    // emit the f16 copy of this thread's 16-feature slice of x[n]
    const int f0 = r * 16;
#pragma unroll
    for (int j = 0; j < 16; ++j)
        Xh[n * FDIM + f0 + j] = (_Float16)xr[f0 + j];
}

// ---------------------------------------------------------------------------
// Kernel 2: fused weighted rule-GEMM.
//   out[n,o] = sum_r fw[n,r] * ( X[n,:] . Wr[:,o] + b[r,o] )
// 256 threads = 8 waves; block covers 64 rows x 128 cols. Wave w owns one
// 16-column slab and four 16x16 D tiles stacked in M, so every B fragment is
// reused 4x (cuts WT L2 traffic 4x vs a 16-row tile). A tile is staged into
// LDS with async-to-LDS copies (ASYNCcnt) when the toolchain exposes them.
// ---------------------------------------------------------------------------
__global__ __launch_bounds__(256) void anfis_gemm_kernel(
    const _Float16* __restrict__ Xh, const _Float16* __restrict__ WT,
    const float* __restrict__ fw, const float* __restrict__ bias,
    float* __restrict__ out)
{
    // padded row stride 136 halves (272 B) -> row bases rotate 4 banks/row,
    // conflict-free ds_load_b128 across the 16 lanes of a fragment read.
    __shared__ _Float16 Ah[MTILE * 136];
    __shared__ float    fwT[RDIM * MTILE];   // transposed: [r][row]

    const int t  = threadIdx.x;
    const int m0 = blockIdx.x * MTILE;

    // --- stage A tile: 64 rows x 128 halves = 16 KB, 4 x 16B per thread ---
#pragma unroll
    for (int j = 0; j < 4; ++j) {
        int id   = t + 256 * j;                // 0..1023 transfers of 16B
        int row  = id >> 4;                    // 0..63
        int colh = (id & 15) * 8;              // 0,8,...,120
        const _Float16* g = Xh + (m0 + row) * FDIM + colh;
        _Float16*       l = &Ah[row * 136 + colh];
#if HAVE_ASYNC_LDS
        __builtin_amdgcn_global_load_async_to_lds_b128(AS1P(g), AS3P(l), 0, 0);
#else
        *(v8h*)l = *(const v8h*)g;
#endif
    }
    // --- stage firing weights transposed: fwT[r*64+row] ---
#pragma unroll
    for (int j = 0; j < 2; ++j) {
        int id  = t + 256 * j;                 // 0..511
        int r   = id >> 6;
        int row = id & 63;
        fwT[id] = fw[(m0 + row) * RDIM + r];
    }
#if HAVE_ASYNC_LDS
    wait_asynccnt0();                          // own wave's async writes done
#endif
    __syncthreads();                           // all waves' LDS writes visible

    const int lane  = t & 31;
    const int wave  = t >> 5;
    const int l16   = lane & 15;
    const int lhalf = lane >> 4;               // 0 / 1
    const int col   = wave * 16 + l16;         // output column of this lane

    v8f accF[4] = {};                          // final weighted accumulators

#pragma unroll
    for (int r = 0; r < RDIM; ++r) {
        v8f acc[4] = {};
#pragma unroll
        for (int kk = 0; kk < 4; ++kk) {
            const int k0 = kk * 32;
            // B operand: lane = column, lane-half selects K 0-15 / 16-31.
            v16h b = *(const v16h*)&WT[(r * ODIM + col) * FDIM + k0 + 16 * lhalf];
#pragma unroll
            for (int mt = 0; mt < 4; ++mt) {
                // A operand, contiguous K layout matching B:
                //   VGPR0-3 : K = 16*lhalf + 0..7
                //   VGPR4-7 : K = 16*lhalf + 8..15  (row = mt*16 + l16)
                const int abase = (mt * 16 + l16) * 136 + k0 + 16 * lhalf;
                v8h alo = *(const v8h*)&Ah[abase];
                v8h ahi = *(const v8h*)&Ah[abase + 8];
                v16h a;
#pragma unroll
                for (int i = 0; i < 8; ++i) { a[i] = alo[i]; a[i + 8] = ahi[i]; }
                acc[mt] = __builtin_amdgcn_wmma_f32_16x16x32_f16(
                    false, a, false, b, (short)0, acc[mt], false, false);
            }
        }
        // weighted accumulate: accF += fw[row,r] * (acc + b[r,col])
        float bv = bias[r * ODIM + col];
        const float* fwr = &fwT[r * MTILE];
#pragma unroll
        for (int mt = 0; mt < 4; ++mt) {
#pragma unroll
            for (int i = 0; i < 8; ++i) {
                // C/D layout: VGPR i <-> row i (lanes 0-15) / i+8 (lanes 16-31)
                float w = fwr[mt * 16 + 8 * lhalf + i];
                accF[mt][i] += w * (acc[mt][i] + bv);
            }
        }
    }

#pragma unroll
    for (int mt = 0; mt < 4; ++mt)
#pragma unroll
        for (int i = 0; i < 8; ++i)
            out[(m0 + mt * 16 + i + 8 * lhalf) * ODIM + col] = accF[mt][i];
}

// ---------------------------------------------------------------------------
// Host-side launcher
// ---------------------------------------------------------------------------
extern "C" void kernel_launch(void* const* d_in, const int* in_sizes, int n_in,
                              void* d_out, int out_size, void* d_ws, size_t ws_size,
                              hipStream_t stream)
{
    const float* x        = (const float*)d_in[0];   // (B,T,F)
    const float* centers  = (const float*)d_in[1];   // (R,F)
    const float* widths   = (const float*)d_in[2];   // (R,F)
    const float* cons_w   = (const float*)d_in[3];   // (R,F,O)
    const float* cons_b   = (const float*)d_in[4];   // (R,O)
    float*       out      = (float*)d_out;           // (B,T,O)

    // workspace layout
    char* ws = (char*)d_ws;
    _Float16* WT = (_Float16*)ws;                                   // 256 KB
    float*    fw = (float*)(ws + RDIM * ODIM * FDIM * sizeof(_Float16));  // 512 KB
    _Float16* Xh = (_Float16*)(ws + RDIM * ODIM * FDIM * sizeof(_Float16)
                                  + N_TOT * RDIM * sizeof(float));  // 4 MB

    anfis_wt_kernel<<<(RDIM * FDIM * ODIM) / 256, 256, 0, stream>>>(cons_w, WT);
    anfis_firing_kernel<<<(N_TOT * RDIM) / 256, 256, 0, stream>>>(
        x, centers, widths, fw, Xh);
    anfis_gemm_kernel<<<N_TOT / MTILE, 256, 0, stream>>>(Xh, WT, fw, cons_b, out);
}
